// SpatialAttnBlock_74929999446754
// MI455X (gfx1250) — matrix operand order repaired and checked by reading, not compiled, not run on hardware
//
#include <hip/hip_runtime.h>
#include <stdint.h>

typedef __attribute__((ext_vector_type(16))) __bf16 v16bf;
typedef __attribute__((ext_vector_type(2)))  __bf16 v2bf;
typedef __attribute__((ext_vector_type(8)))  float  v8f;
typedef __attribute__((ext_vector_type(2)))  float  v2f;

#define HW      4096
#define C_KQ    3
#define D_MODEL 256
#define RSQRT3  0.57735026918962576f

// packed f32x2 -> bf16x2 (RNE); lowers to v_cvt_pk_bf16_f32 on gfx1250
__device__ __forceinline__ uint32_t pack_bf16(float lo, float hi) {
  v2f f; f.x = lo; f.y = hi;
  union { v2bf h; uint32_t u; } c;
  c.h = __builtin_convertvector(f, v2bf);
  return c.u;
}

union ABf16 { uint32_t u[8]; v16bf v; };

// -------------------------------------------------------------------------
// K1: val_bf16[t][o][p] = sum_c Wv[o][c] * v[t][c][p]   (M=256,K=256,N=4096)
// one wave per 16x16 tile; 16384 waves total.
// -------------------------------------------------------------------------
__global__ void __launch_bounds__(256) val_gemm_kernel(
    const float* __restrict__ Wv, const float* __restrict__ v,
    __bf16* __restrict__ val_bf16) {
  const int lane   = threadIdx.x & 31;
  const int wid    = blockIdx.x * 8 + (threadIdx.x >> 5);
  const int n_tile = wid & 255;
  const int m_tile = (wid >> 8) & 15;
  const int t      = wid >> 12;
  const int row    = lane & 15;
  const int hi     = lane >> 4;
  const int col    = n_tile * 16 + row;
  const int m      = m_tile * 16 + row;
  const float* vt  = v + (size_t)t * D_MODEL * HW;

  v8f acc = {};
  for (int kk = 0; kk < D_MODEL; kk += 32) {
    ABf16 a, b;
#pragma unroll
    for (int i = 0; i < 8; ++i) {
      const int kA = kk + (hi ? 8 : 0) + ((i < 4) ? 2 * i : 16 + 2 * (i - 4));
      a.u[i] = pack_bf16(Wv[m * D_MODEL + kA], Wv[m * D_MODEL + kA + 1]);
      const int kB = kk + (hi ? 8 : 0) + ((i < 4) ? 2 * i : 16 + 2 * (i - 4));
      b.u[i] = pack_bf16(vt[(size_t)kB * HW + col],
                         vt[(size_t)(kB + 1) * HW + col]);
    }
    acc = __builtin_amdgcn_wmma_f32_16x16x32_bf16(
        false, a.v, false, b.v, (short)0, acc, false, false);
  }

  __bf16* out = val_bf16 + (size_t)t * D_MODEL * HW;
#pragma unroll
  for (int r = 0; r < 8; ++r) {
    const int mo = m_tile * 16 + (hi ? 8 : 0) + r;
    out[(size_t)mo * HW + col] = (__bf16)acc[r];
  }
}

// -------------------------------------------------------------------------
// K2: scores + softmax over key axis p. Each lane owns one q-column, so the
// softmax reduction is register-local; stores are 128B coalesced. k_t in LDS.
// grid = 4t * 128 blocks, each block covers 32 q-columns, 8 waves split p.
// -------------------------------------------------------------------------
__global__ void __launch_bounds__(256) attn_softmax_kernel(
    const float* __restrict__ kin, const float* __restrict__ qin,
    float* __restrict__ attn) {
  __shared__ float sk[C_KQ * HW];     // 48 KB
  __shared__ float red[256];

  const int tid   = threadIdx.x;
  const int t     = blockIdx.x >> 7;
  const int qbase = (blockIdx.x & 127) * 32;

  for (int i = tid; i < C_KQ * HW; i += 256)
    sk[i] = kin[(size_t)t * C_KQ * HW + i];
  __syncthreads();

  const int wave = tid >> 5, lane = tid & 31;
  const int qcol = qbase + lane;
  const float q0 = qin[((size_t)t * 3 + 0) * HW + qcol] * RSQRT3;
  const float q1 = qin[((size_t)t * 3 + 1) * HW + qcol] * RSQRT3;
  const float q2 = qin[((size_t)t * 3 + 2) * HW + qcol] * RSQRT3;
  const int p0 = wave * 512, p1 = p0 + 512;

  // pass 1: max
  float m = -3.4e38f;
  for (int p = p0; p < p1; ++p) {
    float s = sk[p] * q0 + sk[HW + p] * q1 + sk[2 * HW + p] * q2;
    m = fmaxf(m, s);
  }
  red[wave * 32 + lane] = m;
  __syncthreads();
  float gm = -3.4e38f;
#pragma unroll
  for (int w = 0; w < 8; ++w) gm = fmaxf(gm, red[w * 32 + lane]);
  __syncthreads();

  // pass 2: sum of exp
  float sum = 0.f;
  for (int p = p0; p < p1; ++p) {
    float s = sk[p] * q0 + sk[HW + p] * q1 + sk[2 * HW + p] * q2;
    sum += __expf(s - gm);
  }
  red[wave * 32 + lane] = sum;
  __syncthreads();
  float gs = 0.f;
#pragma unroll
  for (int w = 0; w < 8; ++w) gs += red[w * 32 + lane];
  const float inv = 1.0f / gs;

  // pass 3: normalized write (32 consecutive floats per store)
  float* out = attn + (size_t)t * HW * HW;
  for (int p = p0; p < p1; ++p) {
    float s = sk[p] * q0 + sk[HW + p] * q1 + sk[2 * HW + p] * q2;
    out[(size_t)p * HW + qcol] = __expf(s - gm) * inv;
  }
}

// -------------------------------------------------------------------------
// K3: p_val[t][d][q] = sum_p val[t][d][p] * attn[t][p][q]
// M=256,N=4096,K=4096 per t. One wave handles 4 M-tiles x 1 N-tile (4x B reuse).
// 4096 waves total; 128 K-steps x 4 wmma each.
// -------------------------------------------------------------------------
__global__ void __launch_bounds__(256) pval_gemm_kernel(
    const __bf16* __restrict__ val_bf16, const float* __restrict__ attn,
    float* __restrict__ pval) {
  const int lane   = threadIdx.x & 31;
  const int wid    = blockIdx.x * 8 + (threadIdx.x >> 5);
  const int n_tile = wid & 255;
  const int mg     = (wid >> 8) & 3;
  const int t      = wid >> 10;
  const int row    = lane & 15;
  const int hi     = lane >> 4;
  const int col    = n_tile * 16 + row;
  const int kAoff  = hi ? 8 : 0;
  const int kBoff  = hi ? 8 : 0;

  const float*  At = attn     + (size_t)t * HW * HW;
  const __bf16* Vt = val_bf16 + (size_t)t * D_MODEL * HW;

  v8f acc[4] = {};
  for (int kk = 0; kk < HW; kk += 32) {
    ABf16 b;
#pragma unroll
    for (int i = 0; i < 8; ++i) {
      const int kB = kk + kBoff + ((i < 4) ? 2 * i : 16 + 2 * (i - 4));
      b.u[i] = pack_bf16(At[(size_t)kB * HW + col],
                         At[(size_t)(kB + 1) * HW + col]);
    }
#pragma unroll
    for (int mt = 0; mt < 4; ++mt) {
      const int m = (mg * 4 + mt) * 16 + row;
      const uint32_t* ap = (const uint32_t*)(Vt + (size_t)m * HW + kk);
      ABf16 a;
#pragma unroll
      for (int i = 0; i < 8; ++i) {
        const int kA = kAoff + ((i < 4) ? 2 * i : 16 + 2 * (i - 4));
        a.u[i] = ap[kA >> 1];
      }
      acc[mt] = __builtin_amdgcn_wmma_f32_16x16x32_bf16(
          false, a.v, false, b.v, (short)0, acc[mt], false, false);
    }
  }

  float* out = pval + (size_t)t * D_MODEL * HW;
#pragma unroll
  for (int mt = 0; mt < 4; ++mt) {
#pragma unroll
    for (int r = 0; r < 8; ++r) {
      const int mo = (mg * 4 + mt) * 16 + (hi ? 8 : 0) + r;
      out[(size_t)mo * HW + col] = acc[mt][r];
    }
  }
}

// passthrough copies of k and q
__global__ void __launch_bounds__(256) copy_kq_kernel(
    const float* __restrict__ kin, const float* __restrict__ qin,
    float* __restrict__ outk, float* __restrict__ outq) {
  const int i = blockIdx.x * 256 + threadIdx.x;
  if (i < 49152) { outk[i] = kin[i]; outq[i] = qin[i]; }
}

extern "C" void kernel_launch(void* const* d_in, const int* in_sizes, int n_in,
                              void* d_out, int out_size, void* d_ws, size_t ws_size,
                              hipStream_t stream) {
  (void)in_sizes; (void)n_in; (void)out_size; (void)ws_size;
  const float* k  = (const float*)d_in[0];
  const float* q  = (const float*)d_in[1];
  const float* v  = (const float*)d_in[2];
  const float* Wv = (const float*)d_in[3];

  float* out  = (float*)d_out;
  float* pval = out;                     // 4*256*4096
  float* outk = out + 4194304;           // 49152
  float* outq = outk + 49152;            // 49152
  float* attn = outq + 49152;            // 4*4096*4096

  __bf16* val_bf16 = (__bf16*)d_ws;      // 8.4 MB scratch

  val_gemm_kernel    <<<2048, 256, 0, stream>>>(Wv, v, val_bf16);
  attn_softmax_kernel<<<512, 256, 0, stream>>>(k, q, attn);
  copy_kq_kernel     <<<192, 256, 0, stream>>>(k, q, outk, outq);
  pval_gemm_kernel   <<<512, 256, 0, stream>>>(val_bf16, attn, pval);
}
